// CausalSelfAttention_43052752175115
// MI455X (gfx1250) — matrix operand order repaired and checked by reading, not compiled, not run on hardware
//
#include <hip/hip_runtime.h>
#include <hip/hip_bf16.h>

typedef __attribute__((ext_vector_type(16))) __bf16 v16bf;
typedef __attribute__((ext_vector_type(8)))  __bf16 v8bf;
typedef __attribute__((ext_vector_type(8)))  float  v8f;
typedef unsigned int u32x4 __attribute__((ext_vector_type(4)));
typedef int          i32x4 __attribute__((ext_vector_type(4)));
typedef int          i32x8 __attribute__((ext_vector_type(8)));

#define WMMA_BF16(a, b, c) \
  __builtin_amdgcn_wmma_f32_16x16x32_bf16(false, (a), false, (b), (short)0, (c), false, false)

constexpr int Bc  = 2;
constexpr int Tc  = 2048;
constexpr int Cc  = 1024;
constexpr int NHc = 16;
constexpr int Dc  = 64;

// ---------------------------------------------------------------------------
// f32 -> bf16 conversion
// ---------------------------------------------------------------------------
__global__ void f32_to_bf16_kernel(const float* __restrict__ src,
                                   __bf16* __restrict__ dst, int n) {
  int i = blockIdx.x * blockDim.x + threadIdx.x;
  if (i < n) dst[i] = (__bf16)src[i];
}

// ---------------------------------------------------------------------------
// TDM: issue a 2-D tensor DMA load Global -> LDS.
// D# built per CDNA5 ISA:
//   group0: count=1, lds_addr, global_addr[56:0], type=2
//   group1: data_size=2B, tensor_dim0/1, tile_dim0/1, tensor_dim0_stride
// Dims/strides in data_size (element) units. Groups 2/3 zero (<=2-D tensor).
// ---------------------------------------------------------------------------
__device__ __forceinline__ void tdm_load_2d(unsigned lds_byte_addr,
                                            const void* gptr,
                                            unsigned tile_d0,          // elems/row
                                            unsigned tile_d1,          // rows
                                            unsigned long long stride0 /* elems */) {
  unsigned long long ga = (unsigned long long)(size_t)gptr;
  unsigned long long td0 = 1ull << 20;  // generous tensor_dim0 (no OOB clamp)
  unsigned long long td1 = tile_d1;

  u32x4 g0;
  g0[0] = 1u;                                   // count=1, is_restore=0
  g0[1] = lds_byte_addr;                        // lds_addr
  g0[2] = (unsigned)(ga & 0xffffffffu);         // global_addr[31:0]
  g0[3] = (unsigned)((ga >> 32) & 0x01ffffffu)  // global_addr[56:32]
          | (2u << 30);                         // type=2 ("image")

  i32x8 g1;
  g1[0] = (int)(1u << 16);                      // wg_mask=0, data_size=1 (2B)
  g1[1] = (int)((td0 & 0xffffu) << 16);         // [63:48] = tensor_dim0[15:0]
  g1[2] = (int)(((td0 >> 16) & 0xffffu) |       // [79:64] tensor_dim0[31:16]
                ((td1 & 0xffffu) << 16));       // [95:80] tensor_dim1[15:0]
  g1[3] = (int)(((td1 >> 16) & 0xffffu) |       // [111:96] tensor_dim1[31:16]
                ((tile_d0 & 0xffffu) << 16));   // [127:112] tile_dim0
  g1[4] = (int)(tile_d1 & 0xffffu);             // [143:128] tile_dim1, tile_dim2=0
  g1[5] = (int)(stride0 & 0xffffffffu);         // tensor_dim0_stride[31:0]
  g1[6] = (int)((stride0 >> 32) & 0xffffu);     // tensor_dim0_stride[47:32]
  g1[7] = 0;

  i32x4 z = {0, 0, 0, 0};
#if defined(__clang_major__) && (__clang_major__ >= 23)
  i32x8 z8 = {0, 0, 0, 0, 0, 0, 0, 0};
  __builtin_amdgcn_tensor_load_to_lds(g0, g1, z, z, z8, 0);
#else
  __builtin_amdgcn_tensor_load_to_lds(g0, g1, z, z, 0);
#endif
}

// ---------------------------------------------------------------------------
// Generic  Out[M,N] = A[M,K] @ B[N,K]^T + bias  with WMMA bf16, f32 accum.
// MODE 0: QKV epilogue -> scatter q/k ([B,NH,T,D]) and v transposed
//         ([B,NH,D,T], +bV). MODE 1: f32 store to out[M,N].
// Block: 128 threads (4 waves, 2x2), block tile 64x64, wave tile 32x32.
// ---------------------------------------------------------------------------
template <int MODE>
__global__ __launch_bounds__(128) void wmma_gemm_bf16(
    const __bf16* __restrict__ A, const __bf16* __restrict__ Bm,
    const float* __restrict__ bias, int M, int N, int K,
    __bf16* __restrict__ qO, __bf16* __restrict__ kO, __bf16* __restrict__ vO,
    const float* __restrict__ bV, float* __restrict__ out) {
  const int lane = threadIdx.x & 31;
  const int wave = threadIdx.x >> 5;
  const int h = lane >> 4;   // lane half (ISA 16-bit fragment layout)
  const int c = lane & 15;   // row (A) / column (B,C,D) index within half
  const int m0 = blockIdx.y * 64 + (wave >> 1) * 32;
  const int n0 = blockIdx.x * 64 + (wave & 1) * 32;

  v8f acc[2][2] = {};

  for (int k0 = 0; k0 < K; k0 += 32) {
    v16bf a[2], b[2];
#pragma unroll
    for (int i = 0; i < 2; ++i) {
      // A fragment: lane row m, elems e -> K = 16h+e (lo) and 16h+8+e (hi)
      const __bf16* ap = A + (size_t)(m0 + i * 16 + c) * K + k0;
      v8bf lo = *(const v8bf*)(ap + 16 * h);
      v8bf hi = *(const v8bf*)(ap + 16 * h + 8);
#pragma unroll
      for (int e = 0; e < 8; ++e) { a[i][e] = lo[e]; a[i][8 + e] = hi[e]; }
      // B fragment: lane column n (row of Bm), elems e -> K = 16h+e (contig)
      const __bf16* bp = Bm + (size_t)(n0 + i * 16 + c) * K + k0 + 16 * h;
      b[i] = *(const v16bf*)bp;
    }
#pragma unroll
    for (int i = 0; i < 2; ++i)
#pragma unroll
      for (int j = 0; j < 2; ++j) acc[i][j] = WMMA_BF16(a[i], b[j], acc[i][j]);
  }

  // Epilogue: C/D layout -> lane holds column c, VGPR r holds row 8h+r
#pragma unroll
  for (int i = 0; i < 2; ++i)
#pragma unroll
    for (int j = 0; j < 2; ++j) {
      const int nb = n0 + j * 16 + c;
      const float bia = bias[nb];
#pragma unroll
      for (int r = 0; r < 8; ++r) {
        const int m = m0 + i * 16 + 8 * h + r;
        const float val = acc[i][j][r] + bia;
        if (MODE == 0) {
          const int sec = nb >> 10;       // 0=q 1=k 2=v
          const int cc2 = nb & 1023;
          const int head = cc2 >> 6;
          const int d = cc2 & 63;
          const int bb = m >> 11;         // T = 2048
          const int t = m & 2047;
          if (sec == 0)
            qO[(((size_t)bb * NHc + head) * Tc + t) * Dc + d] = (__bf16)val;
          else if (sec == 1)
            kO[(((size_t)bb * NHc + head) * Tc + t) * Dc + d] = (__bf16)val;
          else
            vO[(((size_t)bb * NHc + head) * Dc + d) * Tc + t] =
                (__bf16)(val + bV[head * Dc + d]);
        } else {
          out[(size_t)m * N + nb] = val;
        }
      }
    }
}

// ---------------------------------------------------------------------------
// Flash attention, TDM-staged K/V.
// Block = 4 waves = 4 consecutive 16-query tiles of one (b, head).
// Per 32-key chunk: wave0/wave1 DMA the K tile (32x64, contiguous 4KB) and
// transposed V tile (64 x 32, row stride T) into double-buffered LDS with
// TENSOR_LOAD_TO_LDS; s_wait_tensorcnt + barrier publishes it; all 4 waves
// then run Q@K^T (WMMA) -> online softmax (shfl butterflies) -> P via LDS
// (C-layout -> A-layout) -> P@V (WMMA) out of LDS. All branches around the
// WMMAs are wave-uniform so EXEC stays all-ones.
// ---------------------------------------------------------------------------
__global__ __launch_bounds__(128) void flash_attn_wmma(
    const __bf16* __restrict__ q, const __bf16* __restrict__ k,
    const __bf16* __restrict__ vT, __bf16* __restrict__ y) {
  // LDS: K tiles 2*4KB | V tiles 2*4KB | P scratch 4*1KB  = 20KB
  __shared__ __align__(128) char smem[2 * 4096 + 2 * 4096 + 4 * 1024];
  const unsigned lds_base = (unsigned)(size_t)(void*)&smem[0];
  const int KOFF0 = 0, VOFF0 = 8192, POFF0 = 16384;

  const int lane = threadIdx.x & 31;
  const int wave = threadIdx.x >> 5;
  const int h = lane >> 4;
  const int c = lane & 15;

  const int QG = Tc / 16 / 4;  // 32 q-tile groups per head
  const int qg = blockIdx.x % QG;
  const int hh = (blockIdx.x / QG) % NHc;
  const int bb = blockIdx.x / (QG * NHc);
  const int qt = qg * 4 + wave;      // this wave's 16-query tile
  const int qrow0 = qt * 16;

  const __bf16* qp = q + ((size_t)bb * NHc + hh) * Tc * Dc;
  const __bf16* kp = k + ((size_t)bb * NHc + hh) * Tc * Dc;
  const __bf16* vp = vT + ((size_t)bb * NHc + hh) * Dc * Tc;

  // Q A-fragments over D = 64 (two 32-wide K steps)
  v16bf aq[2];
#pragma unroll
  for (int i = 0; i < 2; ++i) {
    const __bf16* ap = qp + (size_t)(qrow0 + c) * Dc + i * 32;
    v8bf lo = *(const v8bf*)(ap + 16 * h);
    v8bf hi = *(const v8bf*)(ap + 16 * h + 8);
#pragma unroll
    for (int e = 0; e < 8; ++e) { aq[i][e] = lo[e]; aq[i][8 + e] = hi[e]; }
  }

  float mrun[8], lrun[8];
#pragma unroll
  for (int r = 0; r < 8; ++r) { mrun[r] = -1e30f; lrun[r] = 0.0f; }
  v8f o[4] = {};

  const float scale = 0.125f;  // 1/sqrt(64)
  const int my_nch = (qrow0 + 16 + 31) >> 5;          // my causal chunk count
  const int nch_max = ((qg * 4 + 3) * 16 + 16 + 31) >> 5;  // block max (wave 3)

  // Prologue: stage chunk 0 into buffer 0
  if (wave == 0)
    tdm_load_2d(lds_base + KOFF0, kp + (size_t)0 * Dc, 2048, 1, 2048);
  else if (wave == 1)
    tdm_load_2d(lds_base + VOFF0, vp + 0, 32, 64, (unsigned long long)Tc);

  for (int ch = 0; ch < nch_max; ++ch) {
    const int buf = ch & 1;
    if (wave < 2) __builtin_amdgcn_s_wait_tensorcnt(0);
    __syncthreads();  // staged K/V tile for chunk `ch` visible to all waves

    // Prefetch next chunk into the other buffer (consumed last iteration, before the barrier above).
    if (ch + 1 < nch_max) {
      const int kb1 = (ch + 1) * 32;
      if (wave == 0)
        tdm_load_2d(lds_base + KOFF0 + (1 - buf) * 4096,
                    kp + (size_t)kb1 * Dc, 2048, 1, 2048);
      else if (wave == 1)
        tdm_load_2d(lds_base + VOFF0 + (1 - buf) * 4096, vp + kb1, 32, 64,
                    (unsigned long long)Tc);
    }

    if (ch < my_nch) {  // wave-uniform causal guard
      const int kbase = ch * 32;
      const __bf16* kt = (const __bf16*)(smem + KOFF0 + buf * 4096);
      const __bf16* vt = (const __bf16*)(smem + VOFF0 + buf * 4096);
      __bf16* pt = (__bf16*)(smem + POFF0 + wave * 1024);

      // S = Q @ K^T for 32 keys (two 16-col fragments, K-dim = D)
      v8f s[2];
#pragma unroll
      for (int jj = 0; jj < 2; ++jj) {
        v8f sa = {};
#pragma unroll
        for (int kk = 0; kk < 2; ++kk) {
          v16bf bk = *(const v16bf*)(kt + (jj * 16 + c) * Dc + kk * 32 + 16 * h);
          sa = WMMA_BF16(aq[kk], bk, sa);
        }
        s[jj] = sa;
      }

      // causal mask, scale, online softmax per row (row = qrow0 + 8h + r)
      float alpha[8];
#pragma unroll
      for (int r = 0; r < 8; ++r) {
        const int row = qrow0 + 8 * h + r;
        float v0 = s[0][r] * scale;
        float v1 = s[1][r] * scale;
        if (kbase + c > row) v0 = -1e30f;
        if (kbase + 16 + c > row) v1 = -1e30f;
        float mx = fmaxf(v0, v1);
        mx = fmaxf(mx, __shfl_xor(mx, 1, 32));
        mx = fmaxf(mx, __shfl_xor(mx, 2, 32));
        mx = fmaxf(mx, __shfl_xor(mx, 4, 32));
        mx = fmaxf(mx, __shfl_xor(mx, 8, 32));
        const float mn = fmaxf(mrun[r], mx);
        alpha[r] = __expf(mrun[r] - mn);
        const float p0 = __expf(v0 - mn);
        const float p1 = __expf(v1 - mn);
        float ls = p0 + p1;
        ls += __shfl_xor(ls, 1, 32);
        ls += __shfl_xor(ls, 2, 32);
        ls += __shfl_xor(ls, 4, 32);
        ls += __shfl_xor(ls, 8, 32);
        lrun[r] = lrun[r] * alpha[r] + ls;
        mrun[r] = mn;
        // stash P (bf16) to LDS in [row][key] order for A-layout reload
        pt[(8 * h + r) * 32 + c] = (__bf16)p0;
        pt[(8 * h + r) * 32 + 16 + c] = (__bf16)p1;
      }

      // rescale running output accumulators
#pragma unroll
      for (int t = 0; t < 4; ++t)
#pragma unroll
        for (int r = 0; r < 8; ++r) o[t][r] *= alpha[r];

      // reload P as an A-fragment (16x32, K = 32 keys); same-wave LDS ops
      // are in-order so no barrier needed.
      v16bf apv;
      {
        const __bf16* pp = pt + c * 32;
        v8bf lo = *(const v8bf*)(pp + 16 * h);
        v8bf hi = *(const v8bf*)(pp + 16 * h + 8);
#pragma unroll
        for (int e = 0; e < 8; ++e) { apv[e] = lo[e]; apv[8 + e] = hi[e]; }
      }

      // O += P @ V from the staged transposed V tile (columns contiguous)
#pragma unroll
      for (int t = 0; t < 4; ++t) {
        v16bf bv = *(const v16bf*)(vt + (t * 16 + c) * 32 + 16 * h);
        o[t] = WMMA_BF16(apv, bv, o[t]);
      }
    }
  }

  // normalize and store y[B,T,C] as bf16
#pragma unroll
  for (int r = 0; r < 8; ++r) lrun[r] = 1.0f / lrun[r];
#pragma unroll
  for (int t = 0; t < 4; ++t)
#pragma unroll
    for (int r = 0; r < 8; ++r) {
      const int tq = qrow0 + 8 * h + r;
      const int d = t * 16 + c;
      y[((size_t)bb * Tc + tq) * Cc + hh * Dc + d] = (__bf16)(o[t][r] * lrun[r]);
    }
}

// ---------------------------------------------------------------------------
extern "C" void kernel_launch(void* const* d_in, const int* in_sizes, int n_in,
                              void* d_out, int out_size, void* d_ws,
                              size_t ws_size, hipStream_t stream) {
  const float* x      = (const float*)d_in[0];
  const float* W_attn = (const float*)d_in[1];
  const float* b_attn = (const float*)d_in[2];
  const float* W_proj = (const float*)d_in[3];
  const float* b_proj = (const float*)d_in[4];
  const float* bV     = (const float*)d_in[5];

  const size_t nX = (size_t)Bc * Tc * Cc;          // 4M
  const size_t nWa = (size_t)3 * Cc * Cc;          // 3M
  const size_t nWp = (size_t)Cc * Cc;              // 1M
  const size_t nHead = (size_t)Bc * NHc * Tc * Dc; // 4M

  char* ws = (char*)d_ws;
  __bf16* xb  = (__bf16*)ws; ws += nX * 2;
  __bf16* wab = (__bf16*)ws; ws += nWa * 2;
  __bf16* wpb = (__bf16*)ws; ws += nWp * 2;
  __bf16* qb  = (__bf16*)ws; ws += nHead * 2;
  __bf16* kb  = (__bf16*)ws; ws += nHead * 2;
  __bf16* vb  = (__bf16*)ws; ws += nHead * 2;  // transposed [B,NH,D,T]
  __bf16* yb  = (__bf16*)ws; ws += nX * 2;

  f32_to_bf16_kernel<<<(int)((nX + 255) / 256), 256, 0, stream>>>(x, xb, (int)nX);
  f32_to_bf16_kernel<<<(int)((nWa + 255) / 256), 256, 0, stream>>>(W_attn, wab, (int)nWa);
  f32_to_bf16_kernel<<<(int)((nWp + 255) / 256), 256, 0, stream>>>(W_proj, wpb, (int)nWp);

  dim3 blk(128);
  // QKV: M=4096, N=3072, K=1024
  wmma_gemm_bf16<0><<<dim3(3 * Cc / 64, (Bc * Tc) / 64), blk, 0, stream>>>(
      xb, wab, b_attn, Bc * Tc, 3 * Cc, Cc, qb, kb, vb, bV, nullptr);
  // Attention: B*NH*(T/64) blocks, each = 4 consecutive q-tiles of one head
  flash_attn_wmma<<<dim3(Bc * NHc * (Tc / 64)), blk, 0, stream>>>(qb, kb, vb, yb);
  // Projection: M=4096, N=1024, K=1024 -> f32 out
  wmma_gemm_bf16<1><<<dim3(Cc / 64, (Bc * Tc) / 64), blk, 0, stream>>>(
      yb, wpb, b_proj, Bc * Tc, Cc, Cc, nullptr, nullptr, nullptr, nullptr,
      (float*)d_out);
}
